// SelfAttention_1460288881199
// MI455X (gfx1250) — matrix operand order repaired and checked by reading, not compiled, not run on hardware
//
#include <hip/hip_runtime.h>
#include <hip/hip_bf16.h>

// ---------------------------------------------------------------------------
// Multi-head self-attention forward, CDNA5 (gfx1250), wave32 WMMA f16->f32.
//   B=2, S=2048, E=1024, H=16, D=64.  All GEMM stages on v_wmma_f32_16x16x32_f16.
// ---------------------------------------------------------------------------

typedef __attribute__((ext_vector_type(16))) _Float16 v16h;
typedef __attribute__((ext_vector_type(8)))  _Float16 v8h;
typedef __attribute__((ext_vector_type(8)))  float    v8f;

#define SEQ   2048
#define EMB   1024
#define NHEAD 16
#define HDIM  64
#define MROWS 4096   // B * SEQ

// ---- WMMA fragment loaders (layouts per CDNA5 ISA 7.12.2, 16-bit, wave32) ---

// A-matrix 16x32 f16: lane m = lane&15; lanes<16 k in {0..7,16..23}, lanes>=16
// k in {8..15,24..31}. Row-major source with leading dim `ld` (elements).
__device__ __forceinline__ v16h load_a16(const _Float16* base, int ld) {
    const int lane = threadIdx.x & 31;
    const int m  = lane & 15;
    const int kb = (lane & 16) ? 8 : 0;
    const _Float16* p = base + m * ld + kb;
    v8h lo = *(const v8h*)(p);
    v8h hi = *(const v8h*)(p + 16);
    v16h a;
#pragma unroll
    for (int i = 0; i < 8; ++i) { a[i] = lo[i]; a[i + 8] = hi[i]; }
    return a;
}

// B-matrix 32x16 f16: lane n = lane&15; lanes<16 k in {0..7,16..23}, lanes>=16
// k in {8..15,24..31}. Source layout [n][k] with leading dim `ld` (elements).
__device__ __forceinline__ v16h load_b16(const _Float16* base, int ld) {
    const int lane = threadIdx.x & 31;
    const int n  = lane & 15;
    const int kb = (lane & 16) ? 8 : 0;
    const _Float16* p = base + n * ld + kb;
    v8h lo = *(const v8h*)(p);
    v8h hi = *(const v8h*)(p + 16);
    v16h b;
#pragma unroll
    for (int i = 0; i < 8; ++i) { b[i] = lo[i]; b[i + 8] = hi[i]; }
    return b;
}

__device__ __forceinline__ v8f wmma16(v16h a, v16h b, v8f c) {
    // (neg_a, A, neg_b, B, c_mod, C, reuse_a, reuse_b)
    return __builtin_amdgcn_wmma_f32_16x16x32_f16(false, a, false, b,
                                                  (short)0, c, false, false);
}

// stage a 128x32 f32 tile (row stride EMB) into LDS as f16 [row][32], 256 thr
__device__ __forceinline__ void stage128x32(const float* __restrict__ src,
                                            _Float16* __restrict__ dstLds,
                                            int tid) {
    const int row = tid >> 1;
    const int ko  = (tid & 1) * 16;
    const float4* s = (const float4*)(src + (size_t)row * EMB + ko);
    float4 f0 = s[0], f1 = s[1], f2 = s[2], f3 = s[3];
    _Float16* d = dstLds + row * 32 + ko;
    d[0]  = (_Float16)f0.x; d[1]  = (_Float16)f0.y;
    d[2]  = (_Float16)f0.z; d[3]  = (_Float16)f0.w;
    d[4]  = (_Float16)f1.x; d[5]  = (_Float16)f1.y;
    d[6]  = (_Float16)f1.z; d[7]  = (_Float16)f1.w;
    d[8]  = (_Float16)f2.x; d[9]  = (_Float16)f2.y;
    d[10] = (_Float16)f2.z; d[11] = (_Float16)f2.w;
    d[12] = (_Float16)f3.x; d[13] = (_Float16)f3.y;
    d[14] = (_Float16)f3.z; d[15] = (_Float16)f3.w;
}

// ---------------------------------------------------------------------------
// Kernel 1: Y = X(f32)[4096,1024] @ W(f32)[1024,1024]^T + bias, output f16.
//   vmode==0: dst layout [B,H,S,D]   (for Q, K)
//   vmode==1: dst layout [B,H,D,S]   (transposed, for V -> PV B-fragments)
// Grid (8,32), block 256: 8 waves in 4x2, 32x64 per wave, 128x128 per block.
// ---------------------------------------------------------------------------
__global__ void proj_kernel(const float* __restrict__ x,
                            const float* __restrict__ W,
                            const float* __restrict__ bias,
                            _Float16* __restrict__ dst, int vmode) {
    __shared__ _Float16 As[128 * 32];  // 128 rows of X, k-run 32
    __shared__ _Float16 Bs[128 * 32];  // 128 rows of W (gemm cols), k-run 32

    const int tid  = threadIdx.x;
    const int wave = tid >> 5;
    const int lane = tid & 31;
    const int wy = wave >> 1;          // 0..3  (32 rows each)
    const int wx = wave & 1;           // 0..1  (64 cols each)
    const int m0 = blockIdx.y * 128;
    const int n0 = blockIdx.x * 128;

    v8f c[2][4];
#pragma unroll
    for (int i = 0; i < 2; ++i)
#pragma unroll
        for (int j = 0; j < 4; ++j)
#pragma unroll
            for (int e = 0; e < 8; ++e) c[i][j][e] = 0.0f;

    for (int k0 = 0; k0 < EMB; k0 += 32) {
        stage128x32(x + (size_t)m0 * EMB + k0, As, tid);
        stage128x32(W + (size_t)n0 * EMB + k0, Bs, tid);
        __syncthreads();

        v16h a0 = load_a16(As + (wy * 32 +  0) * 32, 32);
        v16h a1 = load_a16(As + (wy * 32 + 16) * 32, 32);
        v16h b[4];
#pragma unroll
        for (int ns = 0; ns < 4; ++ns)
            b[ns] = load_b16(Bs + (wx * 64 + ns * 16) * 32, 32);
#pragma unroll
        for (int ns = 0; ns < 4; ++ns) {
            c[0][ns] = wmma16(a0, b[ns], c[0][ns]);
            c[1][ns] = wmma16(a1, b[ns], c[1][ns]);
        }
        __syncthreads();
    }

    const int nloc = lane & 15;
    const int hi8  = (lane & 16) ? 8 : 0;
#pragma unroll
    for (int ms = 0; ms < 2; ++ms)
#pragma unroll
        for (int ns = 0; ns < 4; ++ns)
#pragma unroll
            for (int r = 0; r < 8; ++r) {
                const int m = m0 + wy * 32 + ms * 16 + r + hi8;
                const int n = n0 + wx * 64 + ns * 16 + nloc;
                const float val = c[ms][ns][r] + bias[n];
                const int bI = m >> 11, s = m & (SEQ - 1);
                const int hh = n >> 6,  d = n & (HDIM - 1);
                size_t idx;
                if (vmode)  // [B,H,D,S]
                    idx = ((size_t)((bI * NHEAD + hh) * HDIM + d)) * SEQ + s;
                else        // [B,H,S,D]
                    idx = ((size_t)((bI * NHEAD + hh) * SEQ + s)) * HDIM + d;
                dst[idx] = (_Float16)val;
            }
}

// ---------------------------------------------------------------------------
// Kernel 2: flash attention.  Grid (S/64, H, B), block 128 (4 waves).
// Each wave owns 16 query rows; loop over 64-key tiles with online softmax.
//   scores: S = Qh @ Kh^T  (A,B frags straight from global f16, row-major)
//   masked pre-scale per reference, scale = 1/sqrt(1024)
//   P transposed C-layout -> A-layout via wave-private LDS tile
//   O += P @ V  (B frags straight from global V^T [B,H,D,S])
// Next k-tile K/V prefetched (global_prefetch_b8) while current tile computes.
// Output: att[m, h*64+d] f16, m = b*S + q.
// ---------------------------------------------------------------------------
__global__ void attn_kernel(const _Float16* __restrict__ qh,
                            const _Float16* __restrict__ kh,
                            const _Float16* __restrict__ vt,
                            const unsigned char* __restrict__ mask,
                            _Float16* __restrict__ att) {
    const int wave = threadIdx.x >> 5;
    const int lane = threadIdx.x & 31;
    const int h  = blockIdx.y;
    const int bI = blockIdx.z;
    const int q0 = blockIdx.x * 64 + wave * 16;

    const size_t bh = (size_t)(bI * NHEAD + h);
    const _Float16* qb  = qh + (bh * SEQ + q0) * HDIM;
    const _Float16* kbp = kh + bh * SEQ * HDIM;
    const _Float16* vb  = vt + bh * HDIM * SEQ;
    const unsigned char* mb = mask + (size_t)bI * SEQ * SEQ;

    __shared__ _Float16 Ps[4 * 16 * 64];
    _Float16* lp = Ps + wave * 16 * 64;

    // Q fragments (16 x 64) resident for the whole loop
    v16h aq0 = load_a16(qb, HDIM);        // d 0..31
    v16h aq1 = load_a16(qb + 32, HDIM);   // d 32..63

    v8f o[4];
    float mrun[8], lrun[8];
#pragma unroll
    for (int dt = 0; dt < 4; ++dt)
#pragma unroll
        for (int e = 0; e < 8; ++e) o[dt][e] = 0.0f;
#pragma unroll
    for (int r = 0; r < 8; ++r) { mrun[r] = -1e30f; lrun[r] = 0.0f; }

    const int nloc = lane & 15;
    const int hi8  = (lane & 16) ? 8 : 0;
    const float SC = 0.03125f;  // 1/sqrt(1024)

    for (int k0 = 0; k0 < SEQ; k0 += 64) {
        // ---- prefetch next K tile (64x64 f16 = 8KB) and V^T tile ----
        if (k0 + 64 < SEQ) {
            // lane-strided: 32 lanes x 256B = 8KB covered per tile
            const char* pk = (const char*)(kbp + (size_t)(k0 + 64) * HDIM) + lane * 256;
            __builtin_prefetch(pk, 0, 3);
            const char* pv = (const char*)(vb + (size_t)(k0 + 64)) + lane * 2 * SEQ * 2;
            __builtin_prefetch(pv, 0, 3);
        }
        // ---- scores: 4 tiles of 16x16, depth 64 (2 WMMAs each) ----
        v8f s[4];
#pragma unroll
        for (int nt = 0; nt < 4; ++nt) {
            const _Float16* kp = kbp + (size_t)(k0 + nt * 16) * HDIM;
            v16h b0 = load_b16(kp, HDIM);
            v16h b1 = load_b16(kp + 32, HDIM);
            v8f acc;
#pragma unroll
            for (int e = 0; e < 8; ++e) acc[e] = 0.0f;
            acc   = wmma16(aq0, b0, acc);
            s[nt] = wmma16(aq1, b1, acc);
        }
        // ---- mask (before scaling, per reference), then scale ----
#pragma unroll
        for (int nt = 0; nt < 4; ++nt)
#pragma unroll
            for (int r = 0; r < 8; ++r) {
                const int kp = k0 + nt * 16 + nloc;
                const int mg = q0 + r + hi8;
                const float e = mb[(size_t)mg * SEQ + kp] ? s[nt][r] : -1e20f;
                s[nt][r] = e * SC;
            }
        // ---- online softmax (rows live across 16-lane halves) ----
#pragma unroll
        for (int r = 0; r < 8; ++r) {
            float lm = fmaxf(fmaxf(s[0][r], s[1][r]), fmaxf(s[2][r], s[3][r]));
#pragma unroll
            for (int off = 1; off < 16; off <<= 1)
                lm = fmaxf(lm, __shfl_xor(lm, off, 32));
            const float mn   = fmaxf(mrun[r], lm);
            const float corr = __expf(mrun[r] - mn);
            mrun[r] = mn;
            float rs = 0.0f;
#pragma unroll
            for (int nt = 0; nt < 4; ++nt) {
                const float p = __expf(s[nt][r] - mn);
                s[nt][r] = p;
                rs += p;
            }
#pragma unroll
            for (int off = 1; off < 16; off <<= 1)
                rs += __shfl_xor(rs, off, 32);
            lrun[r] = lrun[r] * corr + rs;
#pragma unroll
            for (int dt = 0; dt < 4; ++dt) o[dt][r] *= corr;
        }
        // ---- P (C-layout f32) -> LDS row-major f16 [16 x 64] ----
#pragma unroll
        for (int nt = 0; nt < 4; ++nt)
#pragma unroll
            for (int r = 0; r < 8; ++r)
                lp[(r + hi8) * 64 + nt * 16 + nloc] = (_Float16)s[nt][r];
        __syncthreads();

        // ---- O += P @ V : A from LDS, B from global V^T ----
        v16h p0 = load_a16(lp, 64);        // kpos 0..31
        v16h p1 = load_a16(lp + 32, 64);   // kpos 32..63
#pragma unroll
        for (int dt = 0; dt < 4; ++dt) {
            const _Float16* vp = vb + (size_t)(dt * 16) * SEQ + k0;
            v16h bv0 = load_b16(vp, SEQ);
            v16h bv1 = load_b16(vp + 32, SEQ);
            o[dt] = wmma16(p0, bv0, o[dt]);
            o[dt] = wmma16(p1, bv1, o[dt]);
        }
        __syncthreads();
    }

    // ---- normalize and store att [4096, 1024] f16 ----
#pragma unroll
    for (int dt = 0; dt < 4; ++dt)
#pragma unroll
        for (int r = 0; r < 8; ++r) {
            const float val = o[dt][r] / lrun[r];
            const size_t row = (size_t)(bI * SEQ + q0 + r + hi8);
            att[row * EMB + h * HDIM + dt * 16 + nloc] = (_Float16)val;
        }
}

// ---------------------------------------------------------------------------
// Kernel 3: out = att(f16)[4096,1024] @ Wo(f32)^T + bo, output f32.
// A fragments direct from global f16; Wo staged f32->f16 in LDS.
// Grid (8,32), block 256: 8 waves 4x2, 32x64 per wave, 128x128 per block.
// ---------------------------------------------------------------------------
__global__ void outproj_kernel(const _Float16* __restrict__ att,
                               const float* __restrict__ Wo,
                               const float* __restrict__ bo,
                               float* __restrict__ out) {
    __shared__ _Float16 Bs[128 * 32];

    const int tid  = threadIdx.x;
    const int wave = tid >> 5;
    const int lane = tid & 31;
    const int wy = wave >> 1;
    const int wx = wave & 1;
    const int m0 = blockIdx.y * 128;
    const int n0 = blockIdx.x * 128;

    v8f c[2][4];
#pragma unroll
    for (int i = 0; i < 2; ++i)
#pragma unroll
        for (int j = 0; j < 4; ++j)
#pragma unroll
            for (int e = 0; e < 8; ++e) c[i][j][e] = 0.0f;

    for (int k0 = 0; k0 < EMB; k0 += 32) {
        stage128x32(Wo + (size_t)n0 * EMB + k0, Bs, tid);
        __syncthreads();

        v16h a0 = load_a16(att + (size_t)(m0 + wy * 32) * EMB + k0, EMB);
        v16h a1 = load_a16(att + (size_t)(m0 + wy * 32 + 16) * EMB + k0, EMB);
        v16h b[4];
#pragma unroll
        for (int ns = 0; ns < 4; ++ns)
            b[ns] = load_b16(Bs + (wx * 64 + ns * 16) * 32, 32);
#pragma unroll
        for (int ns = 0; ns < 4; ++ns) {
            c[0][ns] = wmma16(a0, b[ns], c[0][ns]);
            c[1][ns] = wmma16(a1, b[ns], c[1][ns]);
        }
        __syncthreads();
    }

    const int nloc = lane & 15;
    const int hi8  = (lane & 16) ? 8 : 0;
#pragma unroll
    for (int ms = 0; ms < 2; ++ms)
#pragma unroll
        for (int ns = 0; ns < 4; ++ns)
#pragma unroll
            for (int r = 0; r < 8; ++r) {
                const int m = m0 + wy * 32 + ms * 16 + r + hi8;
                const int n = n0 + wx * 64 + ns * 16 + nloc;
                out[(size_t)m * EMB + n] = c[ms][ns][r] + bo[n];
            }
}

// ---------------------------------------------------------------------------
extern "C" void kernel_launch(void* const* d_in, const int* in_sizes, int n_in,
                              void* d_out, int out_size, void* d_ws, size_t ws_size,
                              hipStream_t stream) {
    const float* v  = (const float*)d_in[0];
    const float* k  = (const float*)d_in[1];
    const float* q  = (const float*)d_in[2];
    const unsigned char* mask = (const unsigned char*)d_in[3];
    const float* Wv = (const float*)d_in[4];
    const float* bv = (const float*)d_in[5];
    const float* Wk = (const float*)d_in[6];
    const float* bk = (const float*)d_in[7];
    const float* Wq = (const float*)d_in[8];
    const float* bq = (const float*)d_in[9];
    const float* Wo = (const float*)d_in[10];
    const float* bo = (const float*)d_in[11];
    float* out = (float*)d_out;

    // workspace: qh, kh (both [B,H,S,D] f16), vt ([B,H,D,S] f16), att ([4096,1024] f16)
    _Float16* qh  = (_Float16*)d_ws;
    _Float16* kh  = qh + (size_t)MROWS * EMB;
    _Float16* vt  = kh + (size_t)MROWS * EMB;
    _Float16* att = vt + (size_t)MROWS * EMB;

    const dim3 gp(EMB / 128, MROWS / 128);
    const dim3 bp(256);
    proj_kernel<<<gp, bp, 0, stream>>>(q, Wq, bq, qh, 0);
    proj_kernel<<<gp, bp, 0, stream>>>(k, Wk, bk, kh, 0);
    proj_kernel<<<gp, bp, 0, stream>>>(v, Wv, bv, vt, 1);

    attn_kernel<<<dim3(SEQ / 64, NHEAD, 2), dim3(128), 0, stream>>>(qh, kh, vt, mask, att);

    outproj_kernel<<<gp, bp, 0, stream>>>(att, Wo, bo, out);
}
